// VectorTransformerBlock_14697378087033
// MI455X (gfx1250) — matrix operand order repaired and checked by reading, not compiled, not run on hardware
//
#include <hip/hip_runtime.h>
#include <hip/hip_bf16.h>
#include <math.h>

#define SEQ 4096
#define EMB 1024

typedef __attribute__((ext_vector_type(16))) __bf16        v16bf;
typedef __attribute__((ext_vector_type(8)))  float         v8f;
typedef __attribute__((ext_vector_type(4)))  unsigned int  v4u;

#define WMMA_BF16(A, B, C) \
  __builtin_amdgcn_wmma_f32_16x16x32_bf16(false, (A), false, (B), (short)0, (C), false, false)

// ---- operand loaders from LDS (row-major [64 rows][32 k] bf16 tiles) ----
// A 16x32 bf16 layout: lane -> M = lane%16, h = lane/16; K chunks [8h,8h+8) and [16+8h,16+8h+8)
__device__ __forceinline__ v16bf ld_a(const __bf16* base, int row, int h) {
  union { v16bf v; v4u q[2]; } u;
  const __bf16* p = base + row * 32;
  u.q[0] = *(const v4u*)(p + 8 * h);
  u.q[1] = *(const v4u*)(p + 16 + 8 * h);
  return u.v;
}
// B 32x16 bf16 layout: lane -> N = lane%16, h = lane/16; K chunks [8h,8h+8) and [16+8h,16+8h+8)
__device__ __forceinline__ v16bf ld_b(const __bf16* base, int col, int h) {
  union { v16bf v; v4u q[2]; } u;
  const __bf16* p = base + col * 32;
  u.q[0] = *(const v4u*)(p + 8 * h);
  u.q[1] = *(const v4u*)(p + 16 + 8 * h);
  return u.v;
}

// Load one K-step of A rows / B cols into registers (global -> VGPR).
__device__ __forceinline__ void load_regs(const float* __restrict__ X, int rb, int cb, int k0,
                                          int tid, float* fa, float* fb) {
  const int row = tid >> 2;          // 0..63
  const int kk  = (tid & 3) << 3;    // 0,8,16,24
  const float* pa = X + (size_t)(rb + row) * EMB + k0 + kk;
  const float* pb = X + (size_t)(cb + row) * EMB + k0 + kk;
  *(float4*)&fa[0] = ((const float4*)pa)[0];
  *(float4*)&fa[4] = ((const float4*)pa)[1];
  *(float4*)&fb[0] = ((const float4*)pb)[0];
  *(float4*)&fb[4] = ((const float4*)pb)[1];
}

// Convert f32 regs -> bf16 hi/lo split and store to the current LDS buffer.
__device__ __forceinline__ void store_tiles(const float* fa, const float* fb,
                                            __bf16* Ahi, __bf16* Alo, __bf16* Bhi, __bf16* Blo,
                                            int tid) {
  const int row = tid >> 2;
  const int kk  = (tid & 3) << 3;
#pragma unroll
  for (int i = 0; i < 8; ++i) {
    float a = fa[i]; __bf16 ah = (__bf16)a;
    Ahi[row * 32 + kk + i] = ah;
    Alo[row * 32 + kk + i] = (__bf16)(a - (float)ah);
    float b = fb[i]; __bf16 bh = (__bf16)b;
    Bhi[row * 32 + kk + i] = bh;
    Blo[row * 32 + kk + i] = (__bf16)(b - (float)bh);
  }
}

// Compute 64x64 score block (this wave's two 16x16 tiles) with bf16x2-split WMMA, K=1024.
// Software-pipelined: double-buffered LDS, next K-step's global loads issued under WMMAs,
// one barrier per K-step (backend inserts s_wait_dscnt before s_barrier for LDS ordering).
__device__ __forceinline__ void score_block(const float* __restrict__ X, int rb, int cb,
                                            __bf16* Ahi, __bf16* Alo, __bf16* Bhi, __bf16* Blo,
                                            int tid, int rt, int chh, int ln, int h,
                                            v8f& c0, v8f& c1) {
  float fa[8], fb[8];
  load_regs(X, rb, cb, 0, tid, fa, fb);
  v8f z = {};
  c0 = z; c1 = z;
  int cur = 0;
  for (int k0 = 0; k0 < EMB; k0 += 32) {
    const int off = cur * (64 * 32);
    store_tiles(fa, fb, Ahi + off, Alo + off, Bhi + off, Blo + off, tid);
    __syncthreads();
    if (k0 + 32 < EMB) load_regs(X, rb, cb, k0 + 32, tid, fa, fb);  // prefetch next step
    v16bf ah  = ld_a(Ahi + off, rt * 16 + ln, h);
    v16bf al  = ld_a(Alo + off, rt * 16 + ln, h);
    v16bf b0h = ld_b(Bhi + off, chh * 32 + ln, h);
    v16bf b0l = ld_b(Blo + off, chh * 32 + ln, h);
    v16bf b1h = ld_b(Bhi + off, chh * 32 + 16 + ln, h);
    v16bf b1l = ld_b(Blo + off, chh * 32 + 16 + ln, h);
    c0 = WMMA_BF16(ah, b0h, c0);
    c0 = WMMA_BF16(ah, b0l, c0);
    c0 = WMMA_BF16(al, b0h, c0);
    c1 = WMMA_BF16(ah, b1h, c1);
    c1 = WMMA_BF16(ah, b1l, c1);
    c1 = WMMA_BF16(al, b1h, c1);
    cur ^= 1;
  }
}

// w[i] = sum_j mvec[j] * softmax_rows(X X^T + mask)[j][i], accumulated with atomics.
__global__ __launch_bounds__(256) void attn_colsum_kernel(const float* __restrict__ X,
                                                          const float* __restrict__ mask,
                                                          const float* __restrict__ mvec,
                                                          float* __restrict__ w) {
  __shared__ __align__(16) __bf16 Ahi[2 * 64 * 32], Alo[2 * 64 * 32];
  __shared__ __align__(16) __bf16 Bhi[2 * 64 * 32], Blo[2 * 64 * 32];
  __shared__ float sM[2][64], sL[2][64], wblk[64];
  const int tid  = threadIdx.x;
  const int lane = tid & 31;
  const int wv   = tid >> 5;
  const int ln   = lane & 15, h = lane >> 4;
  const int rt   = wv >> 1, chh = wv & 1;   // row tile 0..3, column half 0..1
  const int rb   = blockIdx.x * 64;

  float rmax[8], rsum[8];
#pragma unroll
  for (int v = 0; v < 8; ++v) { rmax[v] = -1e30f; rsum[v] = 0.f; }

  v8f c0, c1;
  // ---------------- PASS 1: running row max / sum (online softmax stats) ----------------
  for (int cb = 0; cb < SEQ; cb += 64) {
    if (mask && cb + 64 < SEQ) {   // prefetch next column block of the mask
      size_t pnext = (size_t)(rb + rt * 16 + 8 * h) * SEQ + (size_t)(cb + 64 + chh * 32 + ln);
#pragma unroll
      for (int v = 0; v < 8; ++v) __builtin_prefetch(&mask[pnext + (size_t)v * SEQ], 0, 1);
    }
    score_block(X, rb, cb, Ahi, Alo, Bhi, Blo, tid, rt, chh, ln, h, c0, c1);
    if (mask) {
      size_t mbase = (size_t)(rb + rt * 16 + 8 * h) * SEQ + (size_t)(cb + chh * 32 + ln);
#pragma unroll
      for (int v = 0; v < 8; ++v) {
        c0[v] += mask[mbase + (size_t)v * SEQ];
        c1[v] += mask[mbase + (size_t)v * SEQ + 16];
      }
    }
#pragma unroll
    for (int v = 0; v < 8; ++v) {
      float t = fmaxf(c0[v], c1[v]);
      t = fmaxf(t, __shfl_xor(t, 1));
      t = fmaxf(t, __shfl_xor(t, 2));
      t = fmaxf(t, __shfl_xor(t, 4));
      t = fmaxf(t, __shfl_xor(t, 8));
      float mn = fmaxf(rmax[v], t);
      float s  = __expf(c0[v] - mn) + __expf(c1[v] - mn);
      s += __shfl_xor(s, 1);
      s += __shfl_xor(s, 2);
      s += __shfl_xor(s, 4);
      s += __shfl_xor(s, 8);
      rsum[v] = rsum[v] * __expf(rmax[v] - mn) + s;
      rmax[v] = mn;
    }
  }
  // merge the two column-half waves per row
  if (ln == 0) {
#pragma unroll
    for (int v = 0; v < 8; ++v) {
      sM[chh][rt * 16 + 8 * h + v] = rmax[v];
      sL[chh][rt * 16 + 8 * h + v] = rsum[v];
    }
  }
  __syncthreads();
  if (tid < 64) {
    float ma = sM[0][tid], mb = sM[1][tid];
    float M  = fmaxf(ma, mb);
    float L  = sL[0][tid] * __expf(ma - M) + sL[1][tid] * __expf(mb - M);
    sM[0][tid] = M;
    sL[0][tid] = 1.0f / L;
  }
  __syncthreads();
  float Ms[8], wrl[8];
#pragma unroll
  for (int v = 0; v < 8; ++v) {
    Ms[v]  = sM[0][rt * 16 + 8 * h + v];
    wrl[v] = mvec[rb + rt * 16 + 8 * h + v] * sL[0][rt * 16 + 8 * h + v];
  }
  // ---------------- PASS 2: weighted column sums of the softmax ----------------
  for (int cb = 0; cb < SEQ; cb += 64) {
    if (tid < 64) wblk[tid] = 0.f;   // protected by first barrier inside score_block
    if (mask && cb + 64 < SEQ) {
      size_t pnext = (size_t)(rb + rt * 16 + 8 * h) * SEQ + (size_t)(cb + 64 + chh * 32 + ln);
#pragma unroll
      for (int v = 0; v < 8; ++v) __builtin_prefetch(&mask[pnext + (size_t)v * SEQ], 0, 1);
    }
    score_block(X, rb, cb, Ahi, Alo, Bhi, Blo, tid, rt, chh, ln, h, c0, c1);
    if (mask) {
      size_t mbase = (size_t)(rb + rt * 16 + 8 * h) * SEQ + (size_t)(cb + chh * 32 + ln);
#pragma unroll
      for (int v = 0; v < 8; ++v) {
        c0[v] += mask[mbase + (size_t)v * SEQ];
        c1[v] += mask[mbase + (size_t)v * SEQ + 16];
      }
    }
    float t0 = 0.f, t1 = 0.f;
#pragma unroll
    for (int v = 0; v < 8; ++v) {
      t0 += wrl[v] * __expf(c0[v] - Ms[v]);
      t1 += wrl[v] * __expf(c1[v] - Ms[v]);
    }
    t0 += __shfl_xor(t0, 16);        // combine the two row halves (same column)
    t1 += __shfl_xor(t1, 16);
    if (h == 0) {
      atomicAdd(&wblk[chh * 32 + ln], t0);
      atomicAdd(&wblk[chh * 32 + 16 + ln], t1);
    }
    __syncthreads();
    if (tid < 64) atomicAdd(&w[cb + tid], wblk[tid]);
    __syncthreads();
  }
}

__global__ void zero_kernel(float* p, int n) {
  int i = blockIdx.x * blockDim.x + threadIdx.x;
  if (i < n) p[i] = 0.f;
}

// u[e] = sum_j X[j][e] * w[j]
__global__ void xtw_kernel(const float* __restrict__ X, const float* __restrict__ w,
                           float* __restrict__ u) {
  __shared__ float ws[256];
  int e = blockIdx.x * 256 + threadIdx.x;
  float acc = 0.f;
  for (int jb = 0; jb < SEQ; jb += 256) {
    ws[threadIdx.x] = w[jb + threadIdx.x];
    __syncthreads();
    for (int j = 0; j < 256; ++j) acc += X[(size_t)(jb + j) * EMB + e] * ws[j];
    __syncthreads();
  }
  u[e] = acc;
}

__global__ void sumw_kernel(const float* __restrict__ w, float* __restrict__ sumw) {
  const float* p = w + blockIdx.x * SEQ;
  float a = 0.f;
  for (int i = threadIdx.x; i < SEQ; i += 256) a += p[i];
  for (int o = 16; o; o >>= 1) a += __shfl_xor(a, o);
  __shared__ float red[8];
  if ((threadIdx.x & 31) == 0) red[threadIdx.x >> 5] = a;
  __syncthreads();
  if (threadIdx.x == 0) {
    float s = 0.f;
    for (int i = 0; i < 8; ++i) s += red[i];
    sumw[blockIdx.x] = s;
  }
}

// vec[r] = dot(W[r,:], u) + b[r] * sumw   (one wave per output row)
__global__ void wu_kernel(const float* __restrict__ Wm, const float* __restrict__ bm,
                          const float* __restrict__ u, const float* __restrict__ sumw,
                          float* __restrict__ vec) {
  int lane = threadIdx.x & 31;
  int r    = blockIdx.x * 8 + (threadIdx.x >> 5);
  float acc = 0.f;
  for (int e = lane; e < EMB; e += 32) acc += Wm[(size_t)r * EMB + e] * u[e];
  for (int o = 16; o; o >>= 1) acc += __shfl_xor(acc, o);
  if (lane == 0) vec[r] = acc + bm[r] * sumw[0];
}

// out[j] = (vecD - vec_j) / (||vecD - vec_j|| + 1e-8), j in {Q,K,V}
__global__ void finalize_kernel(const float* __restrict__ vec, float* __restrict__ out) {
  const float* vD = vec + 3 * EMB;
  const float* vx = vec + blockIdx.x * EMB;
  __shared__ float red[8];
  __shared__ float sscale;
  int tid = threadIdx.x, lane = tid & 31, wv = tid >> 5;
  float d[4]; float ss = 0.f;
#pragma unroll
  for (int i = 0; i < 4; ++i) {
    int e = tid + 256 * i;
    float df = vD[e] - vx[e];
    d[i] = df; ss += df * df;
  }
  for (int o = 16; o; o >>= 1) ss += __shfl_xor(ss, o);
  if (lane == 0) red[wv] = ss;
  __syncthreads();
  if (tid == 0) {
    float s = 0.f;
    for (int i = 0; i < 8; ++i) s += red[i];
    sscale = 1.0f / (sqrtf(s) + 1e-8f);
  }
  __syncthreads();
#pragma unroll
  for (int i = 0; i < 4; ++i) {
    int e = tid + 256 * i;
    out[blockIdx.x * EMB + e] = d[i] * sscale;
  }
}

extern "C" void kernel_launch(void* const* d_in, const int* in_sizes, int n_in,
                              void* d_out, int out_size, void* d_ws, size_t ws_size,
                              hipStream_t stream) {
  (void)in_sizes; (void)n_in; (void)out_size; (void)ws_size;
  const float* Q     = (const float*)d_in[0];
  const float* K     = (const float*)d_in[1];
  const float* V     = (const float*)d_in[2];
  const float* D     = (const float*)d_in[3];
  const float* mask1 = (const float*)d_in[4];
  const float* mask2 = (const float*)d_in[5];
  const float* Wm[4] = {(const float*)d_in[6], (const float*)d_in[8],
                        (const float*)d_in[10], (const float*)d_in[12]};
  const float* bm[4] = {(const float*)d_in[7], (const float*)d_in[9],
                        (const float*)d_in[11], (const float*)d_in[13]};
  const float* mv[4] = {(const float*)d_in[14], (const float*)d_in[15],
                        (const float*)d_in[16], (const float*)d_in[17]};
  const float* Xs[4]    = {Q, K, V, D};
  const float* masks[4] = {nullptr, mask1, mask2, nullptr};

  float* ws   = (float*)d_ws;
  float* w    = ws;                 // 4 * 4096
  float* u    = w + 4 * SEQ;        // 4 * 1024
  float* vec  = u + 4 * EMB;        // 4 * 1024
  float* sumw = vec + 4 * EMB;      // 4

  zero_kernel<<<(4 * SEQ + 255) / 256, 256, 0, stream>>>(w, 4 * SEQ);
  for (int i = 0; i < 4; ++i)
    attn_colsum_kernel<<<SEQ / 64, 256, 0, stream>>>(Xs[i], masks[i], mv[i], w + i * SEQ);
  for (int i = 0; i < 4; ++i)
    xtw_kernel<<<EMB / 256, 256, 0, stream>>>(Xs[i], w + i * SEQ, u + i * EMB);
  sumw_kernel<<<4, 256, 0, stream>>>(w, sumw);
  for (int i = 0; i < 4; ++i)
    wu_kernel<<<EMB / 8, 256, 0, stream>>>(Wm[i], bm[i], u + i * EMB, sumw + i, vec + i * EMB);
  finalize_kernel<<<3, 256, 0, stream>>>(vec, (float*)d_out);
}
